// FlowEmbedding_28398323761382
// MI455X (gfx1250) — matrix operand order repaired and checked by reading, not compiled, not run on hardware
//
#include <hip/hip_runtime.h>

typedef _Float16 half_t;
typedef _Float16 v16h __attribute__((ext_vector_type(16)));
typedef _Float16 h8   __attribute__((ext_vector_type(8)));
typedef float    v8f  __attribute__((ext_vector_type(8)));

#define NPTS 4096
#define CDIM 128
#define KNN  32
#define KP1  288          // padded input dim for layer 1 (9 * 32)
#define XS   296          // LDS row stride (halves) for x buffer (bank padding)
#define AS   136          // LDS row stride (halves) for activation buffer
#define NWG  8192         // stage-kernel workgroups (2 queries each, 16384 queries)
#define MTOT 524288.0f    // B*S*K rows for batch-norm statistics

// ---------- WMMA fragment loaders (CDNA5 16-bit layouts, wave32) ----------
// A 16x32 f16: lane l -> row m=l%16, hi=l/16; halves 0..7 = K 8hi..8hi+7,
// halves 8..15 = K 8hi+16..8hi+23 (per ISA 7.12.2).
__device__ __forceinline__ v16h load_afrag(const half_t* p) {
    h8 lo = *(const h8*)p;
    h8 hi = *(const h8*)(p + 16);
    v16h a;
#pragma unroll
    for (int i = 0; i < 8; ++i) { a[i] = lo[i]; a[i + 8] = hi[i]; }
    return a;
}
// B 32x16 f16: lane l -> col n=l%16, hi=l/16; halves 0..7 = K 8hi..8hi+7,
// halves 8..15 = K 8hi+16..8hi+23 (same order as A).
__device__ __forceinline__ v16h load_bfrag(const half_t* p) {
    h8 lo = *(const h8*)p;
    h8 hi = *(const h8*)(p + 16);
    v16h b;
#pragma unroll
    for (int i = 0; i < 8; ++i) { b[i] = lo[i]; b[i + 8] = hi[i]; }
    return b;
}

// One wave computes a 16(M) x 128(N) tile: 8 WMMA n-tiles, kdim/32 k-steps.
__device__ __forceinline__ void gemm_wave(const half_t* Abase, int astride, int kdim,
                                          const half_t* Wg, int wstride,
                                          int lane, v8f* acc) {
    const int m  = lane & 15;
    const int hi = lane >> 4;
    v8f zero = {};
#pragma unroll
    for (int nt = 0; nt < 8; ++nt) acc[nt] = zero;
    for (int ks = 0; ks < kdim / 32; ++ks) {
        v16h a = load_afrag(Abase + m * astride + ks * 32 + hi * 8);
#pragma unroll
        for (int nt = 0; nt < 8; ++nt) {
            v16h b = load_bfrag(Wg + (nt * 16 + m) * wstride + ks * 32 + hi * 8);
            acc[nt] = __builtin_amdgcn_wmma_f32_16x16x32_f16(
                false, a, false, b, (short)0, acc[nt], false, false);
        }
    }
}

// Deterministic per-channel sum / sum-of-squares accumulation into per-wave LDS slots.
__device__ __forceinline__ void stats_acc(const v8f* acc, int lane, int wv,
                                          float* sSumW, float* sSqW) {
#pragma unroll
    for (int nt = 0; nt < 8; ++nt) {
        float s = 0.f, q = 0.f;
#pragma unroll
        for (int r = 0; r < 8; ++r) { float v = acc[nt][r]; s += v; q += v * v; }
        s += __shfl_xor(s, 16);   // combine rows of lane halves (same column)
        q += __shfl_xor(q, 16);
        if (lane < 16) {
            int c = nt * 16 + lane;
            sSumW[wv * 128 + c] += s;
            sSqW[wv * 128 + c]  += q;
        }
    }
}

// Apply folded BN affine + ReLU, store f16 activations (D layout: VGPR r -> row r+8*hi).
__device__ __forceinline__ void bn_relu_store(const v8f* acc, const float* ss,
                                              int lane, int rowbase, half_t* actb) {
    const int hi = lane >> 4;
    const int n0 = lane & 15;
#pragma unroll
    for (int nt = 0; nt < 8; ++nt) {
        int c = nt * 16 + n0;
        float sc = ss[c], sh = ss[128 + c];
#pragma unroll
        for (int r = 0; r < 8; ++r) {
            float v = sc * acc[nt][r] + sh;
            v = v > 0.f ? v : 0.f;
            actb[(rowbase + r + 8 * hi) * AS + c] = (half_t)v;
        }
    }
}

// ---------- KNN: top-32 nearest neighbors per query ----------
__global__ __launch_bounds__(128) void knn_kernel(const float* __restrict__ p1,
                                                  const float* __restrict__ p2,
                                                  int* __restrict__ idx) {
    __shared__ float lx[1024], ly[1024], lz[1024];
    __shared__ float listD[128 * 33];   // stride 33: conflict-free per-thread lists
    __shared__ int   listI[128 * 33];
    const int t  = threadIdx.x;
    const int gq = blockIdx.x * 128 + t;        // global query (b*4096 + s)
    const int b  = gq >> 12;
    const int s  = gq & (NPTS - 1);
    const float* pb1 = p1 + (size_t)b * 3 * NPTS;
    const float* pb2 = p2 + (size_t)b * 3 * NPTS;
    const float px = pb1[s], py = pb1[NPTS + s], pz = pb1[2 * NPTS + s];
#pragma unroll
    for (int j = 0; j < KNN; ++j) listD[t * 33 + j] = 1e30f;
    float worst = 1e30f; int wi = 0;
    for (int c0 = 0; c0 < NPTS; c0 += 1024) {
        __syncthreads();
        for (int i = t; i < 1024; i += 128) {
            lx[i] = pb2[c0 + i];
            ly[i] = pb2[NPTS + c0 + i];
            lz[i] = pb2[2 * NPTS + c0 + i];
        }
        __syncthreads();
        for (int i = 0; i < 1024; ++i) {
            float dx = lx[i] - px, dy = ly[i] - py, dz = lz[i] - pz;
            float d = dx * dx + dy * dy + dz * dz;
            if (d < worst) {
                listD[t * 33 + wi] = d;
                listI[t * 33 + wi] = c0 + i;
                float w = -1e30f; int widx = 0;
#pragma unroll
                for (int j = 0; j < KNN; ++j) {
                    float v = listD[t * 33 + j];
                    if (v > w) { w = v; widx = j; }
                }
                worst = w; wi = widx;
            }
        }
    }
#pragma unroll
    for (int j = 0; j < KNN; ++j) idx[(size_t)gq * KNN + j] = listI[t * 33 + j];
}

// ---------- Weight prep: f32 -> f16, W1 columns permuted to [gf, f1, rel, pad] ----------
__global__ void prep_kernel(const float* __restrict__ W1, const float* __restrict__ W2,
                            const float* __restrict__ W3, half_t* __restrict__ W1p,
                            half_t* __restrict__ W2p, half_t* __restrict__ W3p) {
    int i = blockIdx.x * 256 + threadIdx.x;
    if (i < 128 * KP1) {
        int n = i / KP1, j = i % KP1;
        float v = 0.f;
        if (j < 128)      v = W1[n * 259 + 3 + j];            // gathered f2 part
        else if (j < 256) v = W1[n * 259 + 131 + (j - 128)];  // broadcast f1 part
        else if (j < 259) v = W1[n * 259 + (j - 256)];        // rel-coord part
        W1p[i] = (half_t)v;
    } else {
        int i2 = i - 128 * KP1;
        if (i2 < 16384)      W2p[i2]         = (half_t)W2[i2];
        else if (i2 < 32768) W3p[i2 - 16384] = (half_t)W3[i2 - 16384];
    }
}

// ---------- Fused recompute stage kernel ----------
// stage 1..3: recompute layers 1..stage, emit per-WG channel sum/sumsq of last layer.
// stage 4: recompute all layers with folded BN, max over K, write output.
__global__ __launch_bounds__(128) void stage_kernel(
    int stage, const float* __restrict__ p1, const float* __restrict__ p2,
    const float* __restrict__ f1, const float* __restrict__ f2,
    const int* __restrict__ idx, const half_t* __restrict__ W1p,
    const half_t* __restrict__ W2p, const half_t* __restrict__ W3p,
    const float* __restrict__ ss, float* __restrict__ partial, float* __restrict__ out) {
    __shared__ half_t xbuf[64 * XS];     // 64 rows x 288 (padded 296) layer-1 input
    __shared__ half_t act[64 * AS];      // 64 rows x 128 (padded 136) activations
    __shared__ float  sSumW[4 * 128];
    __shared__ float  sSqW[4 * 128];

    const int lane = threadIdx.x & 31;
    const int wv   = threadIdx.x >> 5;       // 4 waves, 16 rows each
    const int q0   = blockIdx.x * 2;         // 2 queries per WG (rows 0..31 / 32..63)
    const int rowbase = wv * 16;

#pragma unroll
    for (int i = 0; i < 4; ++i) { sSumW[wv * 128 + lane + i * 32] = 0.f;
                                  sSqW [wv * 128 + lane + i * 32] = 0.f; }

    // Build this wave's 16 rows of x = [gf(128), f1(128), rel(3), 0...]
    for (int r = 0; r < 16; ++r) {
        int row = rowbase + r;
        int gq  = q0 + (row >> 5);
        int b   = gq >> 12;
        int s   = gq & (NPTS - 1);
        int nb  = idx[(size_t)gq * KNN + (row & 31)];
        const float* f2b = f2 + (size_t)b * CDIM * NPTS;
        const float* f1b = f1 + (size_t)b * CDIM * NPTS;
#pragma unroll
        for (int cc = 0; cc < 4; ++cc) {
            int c = lane + cc * 32;
            xbuf[row * XS + c]       = (half_t)f2b[c * NPTS + nb];
            xbuf[row * XS + 128 + c] = (half_t)f1b[c * NPTS + s];
        }
        float v = 0.f;
        if (lane < 3) {
            v = p2[(size_t)b * 3 * NPTS + lane * NPTS + nb]
              - p1[(size_t)b * 3 * NPTS + lane * NPTS + s];
        }
        xbuf[row * XS + 256 + lane] = (half_t)v;   // rel coords + zero pad to 288
    }

    v8f acc[8];
    bool done = false;

    gemm_wave(&xbuf[rowbase * XS], XS, KP1, W1p, KP1, lane, acc);
    if (stage == 1) { stats_acc(acc, lane, wv, sSumW, sSqW); done = true; }
    if (!done) {
        bn_relu_store(acc, ss + 0, lane, rowbase, act);
        gemm_wave(&act[rowbase * AS], AS, 128, W2p, 128, lane, acc);
        if (stage == 2) { stats_acc(acc, lane, wv, sSumW, sSqW); done = true; }
    }
    if (!done) {
        bn_relu_store(acc, ss + 256, lane, rowbase, act);
        gemm_wave(&act[rowbase * AS], AS, 128, W3p, 128, lane, acc);
        if (stage == 3) { stats_acc(acc, lane, wv, sSumW, sSqW); done = true; }
    }
    if (!done) {
        bn_relu_store(acc, ss + 512, lane, rowbase, act);
        __syncthreads();
        int col = threadIdx.x;                    // 0..127
        for (int q = 0; q < 2; ++q) {
            float m = 0.f;                        // post-ReLU values are >= 0
            for (int r = 0; r < KNN; ++r) {
                float v = (float)act[(q * KNN + r) * AS + col];
                m = v > m ? v : m;
            }
            int gq = q0 + q, b = gq >> 12, s = gq & (NPTS - 1);
            out[((size_t)b * CDIM + col) * NPTS + s] = m;
        }
        return;
    }
    __syncthreads();
    if (threadIdx.x < 128) {
        int c = threadIdx.x;
        partial[(size_t)blockIdx.x * 256 + c] =
            sSumW[c] + sSumW[128 + c] + sSumW[256 + c] + sSumW[384 + c];
        partial[(size_t)blockIdx.x * 256 + 128 + c] =
            sSqW[c] + sSqW[128 + c] + sSqW[256 + c] + sSqW[384 + c];
    }
}

// ---------- Fold batch statistics into per-channel affine (scale, shift) ----------
__global__ __launch_bounds__(256) void stats_kernel(const float* __restrict__ partial,
                                                    const float* __restrict__ g,
                                                    const float* __restrict__ beta,
                                                    float* __restrict__ ssOut) {
    __shared__ float rs[256], rq[256];
    const int c = blockIdx.x, t = threadIdx.x;
    float s = 0.f, q = 0.f;
    for (int w = t; w < NWG; w += 256) {
        s += partial[(size_t)w * 256 + c];
        q += partial[(size_t)w * 256 + 128 + c];
    }
    rs[t] = s; rq[t] = q;
    __syncthreads();
    for (int off = 128; off > 0; off >>= 1) {
        if (t < off) { rs[t] += rs[t + off]; rq[t] += rq[t + off]; }
        __syncthreads();
    }
    if (t == 0) {
        float m  = rs[0] / MTOT;
        float v  = rq[0] / MTOT - m * m;
        float sc = g[c] * rsqrtf(v + 1e-3f);
        ssOut[c]       = sc;
        ssOut[128 + c] = beta[c] - m * sc;   // linear bias cancels under BN -> omitted
    }
}

extern "C" void kernel_launch(void* const* d_in, const int* in_sizes, int n_in,
                              void* d_out, int out_size, void* d_ws, size_t ws_size,
                              hipStream_t stream) {
    const float* points1 = (const float*)d_in[0];
    const float* points2 = (const float*)d_in[1];
    const float* feat1   = (const float*)d_in[2];
    const float* feat2   = (const float*)d_in[3];
    const float* W1      = (const float*)d_in[4];
    const float* g1      = (const float*)d_in[6];
    const float* be1     = (const float*)d_in[7];
    const float* W2      = (const float*)d_in[8];
    const float* g2      = (const float*)d_in[10];
    const float* be2     = (const float*)d_in[11];
    const float* W3      = (const float*)d_in[12];
    const float* g3      = (const float*)d_in[14];
    const float* be3     = (const float*)d_in[15];
    float* out = (float*)d_out;

    char* ws = (char*)d_ws;
    size_t o = 0;
    int*    idx = (int*)(ws + o);    o += (size_t)16384 * KNN * 4;   // 2 MB
    half_t* W1p = (half_t*)(ws + o); o += (size_t)128 * KP1 * 2;     // 72 KB
    half_t* W2p = (half_t*)(ws + o); o += (size_t)128 * 128 * 2;     // 32 KB
    half_t* W3p = (half_t*)(ws + o); o += (size_t)128 * 128 * 2;     // 32 KB
    float*  ss  = (float*)(ws + o);  o += (size_t)3 * 256 * 4;       // folded BN affines
    float*  partial = (float*)(ws + o);                              // 8 MB

    prep_kernel<<<272, 256, 0, stream>>>(W1, W2, W3, W1p, W2p, W3p);
    knn_kernel<<<128, 128, 0, stream>>>(points1, points2, idx);

    stage_kernel<<<NWG, 128, 0, stream>>>(1, points1, points2, feat1, feat2, idx,
                                          W1p, W2p, W3p, ss, partial, out);
    stats_kernel<<<128, 256, 0, stream>>>(partial, g1, be1, ss);
    stage_kernel<<<NWG, 128, 0, stream>>>(2, points1, points2, feat1, feat2, idx,
                                          W1p, W2p, W3p, ss, partial, out);
    stats_kernel<<<128, 256, 0, stream>>>(partial, g2, be2, ss + 256);
    stage_kernel<<<NWG, 128, 0, stream>>>(3, points1, points2, feat1, feat2, idx,
                                          W1p, W2p, W3p, ss, partial, out);
    stats_kernel<<<128, 256, 0, stream>>>(partial, g3, be3, ss + 512);
    stage_kernel<<<NWG, 128, 0, stream>>>(4, points1, points2, feat1, feat2, idx,
                                          W1p, W2p, W3p, ss, partial, out);
}
